// PointerNetAttention_29678224015928
// MI455X (gfx1250) — matrix operand order repaired and checked by reading, not compiled, not run on hardware
//
#include <hip/hip_runtime.h>

#define S_TOT 8192
#define B_TOT 32
#define H_TOT 256
#define K_TOT 256
#define G_S   4                      // s-values per block
#define M_ROWS (G_S * B_TOT)         // 128 rows of Ref_flat per block

typedef __attribute__((ext_vector_type(16))) __bf16 v16bf;
typedef __attribute__((ext_vector_type(8)))  float  v8f;

union Frag { uint4 u[2]; v16bf v; };

__device__ __forceinline__ float fast_tanh(float x) {
#if __has_builtin(__builtin_amdgcn_tanhf)
  return __builtin_amdgcn_tanhf(x);          // v_tanh_f32 on gfx1250
#elif __has_builtin(__builtin_amdgcn_tanh_f32)
  return __builtin_amdgcn_tanh_f32(x);
#else
  return tanhf(x);
#endif
}

// Split fp32 into bf16 hi + bf16 lo (truncation; lo captures the remainder).
__device__ __forceinline__ void split_bf16(float x, unsigned short& hi, unsigned short& lo) {
  unsigned int xb = __float_as_uint(x);
  hi = (unsigned short)(xb >> 16);
  float hf = __uint_as_float(((unsigned int)hi) << 16);
  lo = (unsigned short)(__float_as_uint(x - hf) >> 16);
}

// ---------------- Prologue 1: q = query @ Wq^T + bq  -> ws ----------------
__global__ __launch_bounds__(256)
void pn_qproj(const float* __restrict__ query, const float* __restrict__ Wq,
              const float* __restrict__ bq, float* __restrict__ qout) {
  int b = blockIdx.x;
  int h = threadIdx.x;
  const float4* qr = (const float4*)(query + (b << 8));
  const float4* wr = (const float4*)(Wq + (h << 8));
  float s = 0.f;
#pragma unroll 8
  for (int i = 0; i < 64; ++i) {
    float4 a = qr[i], w = wr[i];
    s += a.x * w.x + a.y * w.y + a.z * w.z + a.w * w.w;
  }
  qout[(b << 8) + h] = s + bq[h];
}

// ---------------- Prologue 2: pack Wr into B-fragment bf16 hi/lo ----------------
// B packing used here (32x16 bf16, contiguous k): lanes 0-15 hold K=0..15 (N=lane),
// lanes 16-31 hold K=16..31 (N=lane-16); each lane stores 16 consecutive K values.
__global__ __launch_bounds__(32)
void pn_wpack(const float* __restrict__ Wr,
              unsigned short* __restrict__ whi, unsigned short* __restrict__ wlo) {
  int tile = blockIdx.x;            // = ntg*8 + ks, 0..127
  int ntg  = tile >> 3;             // h tile (16 h each)
  int ks   = tile & 7;              // k step (32 r each)
  int lane = threadIdx.x;           // 0..31
  int h     = (ntg << 4) + (lane & 15);
  int kbase = (ks << 5) + ((lane >> 4) << 4);
  unsigned short hi[16], lo[16];
#pragma unroll
  for (int p = 0; p < 16; ++p)
    split_bf16(Wr[h * K_TOT + kbase + p], hi[p], lo[p]);
  size_t off = ((size_t)(tile * 32 + lane)) << 4;   // 16 ushort = 32B per lane
#pragma unroll
  for (int p = 0; p < 16; ++p) { whi[off + p] = hi[p]; wlo[off + p] = lo[p]; }
}

// ---------------- Main kernel ----------------
// Block = 8 waves / 256 threads, handles s in [blk*4, blk*4+4), all b, all h.
// Wave w owns N-tiles {2w, 2w+1} (h in [32w, 32w+32)) and loops over all 8
// M-tiles, so weights are loaded exactly once per block from L2 and each wave
// holds all 4 s-values of every (b,h) it produces -> in-register s-transpose
// for the e store (no LDS staging, no per-chunk barriers).
__global__ __launch_bounds__(256)
void pn_main(const float* __restrict__ ref, const float* __restrict__ qws,
             const unsigned short* __restrict__ whi, const unsigned short* __restrict__ wlo,
             const float* __restrict__ br, const float* __restrict__ v,
             float* __restrict__ e_out, float* __restrict__ u_out) {
  __shared__ __align__(16) unsigned short A_hi[8 * 8 * 32 * 16];  // 64 KB
  __shared__ __align__(16) unsigned short A_lo[8 * 8 * 32 * 16];  // 64 KB
  __shared__ float u_sh[B_TOT * G_S];                             // 512 B

  const int t   = threadIdx.x;
  const int blk = blockIdx.x;

  // ---- Stage 128 ref rows -> LDS as A-fragment-ordered bf16 hi/lo ----
  // A packing used here (16x32 bf16, contiguous k): lanes (M=lane&15) hold K {16q..16q+15}
  // with q = lane>>4; position idx = kk&15.
  {
    const float4* src = (const float4*)(ref + (size_t)blk * M_ROWS * K_TOT);
#pragma unroll 4
    for (int i = 0; i < 32; ++i) {               // 8192 float4 / 256 threads
      int fidx = t + (i << 8);
      int m  = fidx >> 6;                        // row 0..127
      int kq = (fidx & 63) << 2;                 // k 0..255, step 4
      float4 x = src[fidx];
      unsigned short h0,h1,h2,h3,l0,l1,l2,l3;
      split_bf16(x.x,h0,l0); split_bf16(x.y,h1,l1);
      split_bf16(x.z,h2,l2); split_bf16(x.w,h3,l3);
      int mt   = m >> 4;
      int ks   = kq >> 5;
      int kk   = kq & 31;
      int half = kk >> 4;
      int idx0 = kk & 15;
      int lane = (m & 15) + (half << 4);
      int off  = (((mt * 8 + ks) * 32 + lane) << 4) + idx0;
      *(ushort4*)&A_hi[off] = make_ushort4(h0,h1,h2,h3);
      *(ushort4*)&A_lo[off] = make_ushort4(l0,l1,l2,l3);
    }
    if (t < B_TOT * G_S) u_sh[t] = 0.0f;
  }
  __syncthreads();

  const int wave = t >> 5;
  const int lane = t & 31;
  const int N    = lane & 15;        // C/D: lane -> column (h)
  const int h0   = wave << 5;        // wave owns h in [32w, 32w+32)

  const v8f zero8 = {0.f,0.f,0.f,0.f,0.f,0.f,0.f,0.f};
  v8f acc[16];                        // [mt*2 + nt], mt = 2*s + b_half
#pragma unroll
  for (int i = 0; i < 16; ++i) acc[i] = zero8;

  for (int ks = 0; ks < 8; ++ks) {
    // Weights for this wave's two N-tiles: loaded once, reused by all 8 M-tiles.
    Frag bh[2], bl[2];
#pragma unroll
    for (int nt = 0; nt < 2; ++nt) {
      int ntg = (wave << 1) + nt;
      size_t wb = ((size_t)((ntg * 8 + ks) * 32 + lane)) << 4;
      bh[nt].u[0] = *(const uint4*)&whi[wb];
      bh[nt].u[1] = *(const uint4*)&whi[wb + 8];
      bl[nt].u[0] = *(const uint4*)&wlo[wb];
      bl[nt].u[1] = *(const uint4*)&wlo[wb + 8];
    }
#pragma unroll
    for (int mt = 0; mt < 8; ++mt) {
      Frag ah, al;
      int abase = ((mt * 8 + ks) * 32 + lane) << 4;
      ah.u[0] = *(const uint4*)&A_hi[abase];
      ah.u[1] = *(const uint4*)&A_hi[abase + 8];
      al.u[0] = *(const uint4*)&A_lo[abase];
      al.u[1] = *(const uint4*)&A_lo[abase + 8];
#pragma unroll
      for (int nt = 0; nt < 2; ++nt) {
        // 2-term split product: hi*hi + hi*lo + lo*hi  (drop lo*lo)
        acc[mt*2+nt] = __builtin_amdgcn_wmma_f32_16x16x32_bf16(false, ah.v, false, bh[nt].v,
                                                               (short)0, acc[mt*2+nt], false, false);
        acc[mt*2+nt] = __builtin_amdgcn_wmma_f32_16x16x32_bf16(false, ah.v, false, bl[nt].v,
                                                               (short)0, acc[mt*2+nt], false, false);
        acc[mt*2+nt] = __builtin_amdgcn_wmma_f32_16x16x32_bf16(false, al.v, false, bh[nt].v,
                                                               (short)0, acc[mt*2+nt], false, false);
      }
    }
  }

  // ---- Epilogue: bias, in-register s-transpose e store, tanh-dot for u ----
  int   hh[2];
  float brv[2], vvv[2];
#pragma unroll
  for (int nt = 0; nt < 2; ++nt) {
    hh[nt]  = h0 + (nt << 4) + N;
    brv[nt] = br[hh[nt]];
    vvv[nt] = v[hh[nt]];
  }

#pragma unroll
  for (int hb = 0; hb < 2; ++hb) {          // b-half (mt parity)
#pragma unroll
    for (int j = 0; j < 8; ++j) {           // C/D VGPR slot -> row within tile
      int M = (lane < 16) ? j : (j + 8);
      int b = (hb << 4) + M;
      float u0 = 0.f, u1 = 0.f, u2 = 0.f, u3 = 0.f;
#pragma unroll
      for (int nt = 0; nt < 2; ++nt) {
        int h = hh[nt];
        float qv = qws[(b << 8) + h];
        // mt = 2*s + hb holds s-th value of this (b,h)
        float e0 = acc[(0 + hb) * 2 + nt][j] + brv[nt];
        float e1 = acc[(2 + hb) * 2 + nt][j] + brv[nt];
        float e2 = acc[(4 + hb) * 2 + nt][j] + brv[nt];
        float e3 = acc[(6 + hb) * 2 + nt][j] + brv[nt];
        float4 ev = make_float4(e0, e1, e2, e3);
        *(float4*)(e_out + (size_t)b * H_TOT * S_TOT
                         + (size_t)h * S_TOT + ((size_t)blk << 2)) = ev;
        u0 += vvv[nt] * fast_tanh(qv + e0);
        u1 += vvv[nt] * fast_tanh(qv + e1);
        u2 += vvv[nt] * fast_tanh(qv + e2);
        u3 += vvv[nt] * fast_tanh(qv + e3);
      }
      // reduce over the 16 h-lanes of each half-wave
#pragma unroll
      for (int off = 1; off < 16; off <<= 1) {
        u0 += __shfl_xor(u0, off, 32);
        u1 += __shfl_xor(u1, off, 32);
        u2 += __shfl_xor(u2, off, 32);
        u3 += __shfl_xor(u3, off, 32);
      }
      if (N == 0) {                         // lanes 0 and 16
        atomicAdd(&u_sh[(b << 2) + 0], u0); // ds_add_f32
        atomicAdd(&u_sh[(b << 2) + 1], u1);
        atomicAdd(&u_sh[(b << 2) + 2], u2);
        atomicAdd(&u_sh[(b << 2) + 3], u3);
      }
    }
  }
  __syncthreads();

  // ---- Final u output: 10 * tanh(u[b,s]) ----
  if (t < B_TOT * G_S) {
    int b  = t >> 2;
    int sl = t & 3;
    u_out[(size_t)b * S_TOT + (blk << 2) + sl] = 10.0f * fast_tanh(u_sh[t]);
  }
}

extern "C" void kernel_launch(void* const* d_in, const int* in_sizes, int n_in,
                              void* d_out, int out_size, void* d_ws, size_t ws_size,
                              hipStream_t stream) {
  (void)in_sizes; (void)n_in; (void)out_size; (void)ws_size;
  const float* query = (const float*)d_in[0];
  const float* ref   = (const float*)d_in[1];
  const float* Wq    = (const float*)d_in[2];
  const float* bq    = (const float*)d_in[3];
  const float* Wr    = (const float*)d_in[4];
  const float* br    = (const float*)d_in[5];
  const float* v     = (const float*)d_in[6];

  float* e_out = (float*)d_out;
  float* u_out = (float*)d_out + (size_t)B_TOT * H_TOT * S_TOT;

  // workspace layout: q [32KB] | Wr_hi packed [128KB] | Wr_lo packed [128KB]
  float*          qws = (float*)d_ws;
  unsigned short* whi = (unsigned short*)((char*)d_ws + 32768);
  unsigned short* wlo = (unsigned short*)((char*)d_ws + 32768 + 131072);

  pn_qproj<<<B_TOT, 256, 0, stream>>>(query, Wq, bq, qws);
  pn_wpack<<<(H_TOT / 16) * (K_TOT / 32), 32, 0, stream>>>(Wr, whi, wlo);
  pn_main<<<S_TOT / G_S, 256, 0, stream>>>(ref, qws, whi, wlo, br, v, e_out, u_out);
}
